// ParallelFcWithAttention_1872605741176
// MI455X (gfx1250) — matrix operand order repaired and checked by reading, not compiled, not run on hardware
//
#include <hip/hip_runtime.h>
#include <hip/hip_bf16.h>

// ---------------------------------------------------------------------------
// ParallelFcWithAttention fused kernel for MI455X (gfx1250, wave32, WMMA).
//
// B=16384, F=10 (padded to 16), D=512, H=8, DH=64.
// One workgroup (4 wave32) per batch element. All matmuls via
// v_wmma_f32_16x16x32_f16 (f16 inputs, f32 accumulate).
// ---------------------------------------------------------------------------

typedef __attribute__((ext_vector_type(16))) _Float16 v16h;
typedef __attribute__((ext_vector_type(8)))  _Float16 v8h;
typedef __attribute__((ext_vector_type(8)))  float    v8f;

#define B_N  16384
#define F_N  10
#define D_N  512
#define H_N  8
#define DH_N 64
#define PS   520   // row stride (halves) for 16 x 512 f16 tiles (bank-spread pad)
#define VTS  24    // Vt row stride (halves), 48B = 16B aligned
#define ATS  24    // attn row stride (halves)
#define OFS  520   // Ofin row stride (floats)

// LDS carve (bytes); Ofin overlays Qs+Ks (dead after attention stage).
#define SM_PS   0
#define SM_QS   16640
#define SM_KS   33280
#define SM_VT   49920
#define SM_OS   74496
#define SM_AT   91136
#define SM_XV   103424
#define SM_MU   103488
#define SM_RS   103552
#define SM_ZP   103616
#define SMEM_BYTES 103680

union V16U { v16h v; v8h h[2]; };

// A fragment (16x32 f16) from an LDS row-major tile.
// Layout: lane l -> row m=l&15; K = (l>>4)*16 + 0..15 (contiguous, matches B).
static __device__ inline v16h lds_loadA(const _Float16* base, int stride, int c0, int lane) {
  const int m = lane & 15, g = lane >> 4;
  const _Float16* p = base + m * stride + c0 + g * 16;
  V16U r;
  r.h[0] = *(const v8h*)(p);
  r.h[1] = *(const v8h*)(p + 8);
  return r.v;
}

// B fragment (32x16 f16) from global weight W[n][k] (row-major, torch layout).
// Layout (contiguous K, same as A): lane l -> column n=l&15; lanes<16 hold K=k0..k0+15,
//             lanes>=16 hold K=k0+16..k0+31 (both contiguous in memory).
static __device__ inline v16h glb_loadB(const _Float16* __restrict__ W, int n0, int k0, int lane) {
  const int n = n0 + (lane & 15), g = lane >> 4;
  const _Float16* p = W + n * D_N + k0 + g * 16;
  V16U r;
  r.h[0] = *(const v8h*)(p);
  r.h[1] = *(const v8h*)(p + 8);
  return r.v;
}

// Same pattern for an LDS tile whose rows play the role of B-columns
// (used for K in scores: B(k=d, n=f2) = K[f2][d], contiguous per lane).
static __device__ inline v16h lds_loadB_rows(const _Float16* base, int stride, int c0, int lane) {
  const int n = lane & 15, g = lane >> 4;
  const _Float16* p = base + n * stride + c0 + g * 16;
  V16U r;
  r.h[0] = *(const v8h*)(p);
  r.h[1] = *(const v8h*)(p + 8);
  return r.v;
}

// ------------------------- weight f32 -> f16 pre-pass ----------------------
__global__ void convert_weights_kernel(const float* __restrict__ Wq,
                                       const float* __restrict__ Wk,
                                       const float* __restrict__ Wv,
                                       const float* __restrict__ Wo,
                                       _Float16* __restrict__ Wh) {
  const int i = blockIdx.x * blockDim.x + threadIdx.x;  // 0 .. D*D-1
  const int N = D_N * D_N;
  Wh[i]         = (_Float16)Wq[i];
  Wh[i + N]     = (_Float16)Wk[i];
  Wh[i + 2 * N] = (_Float16)Wv[i];
  Wh[i + 3 * N] = (_Float16)Wo[i];
}

// ------------------------------ fused kernel -------------------------------
__global__ __launch_bounds__(128, 1) void fused_attn_kernel(
    const float* __restrict__ x, const float* __restrict__ Wf, const float* __restrict__ bf,
    const _Float16* __restrict__ Wh,
    const float* __restrict__ bq, const float* __restrict__ bk, const float* __restrict__ bv,
    const float* __restrict__ bo, const float* __restrict__ gamma, const float* __restrict__ beta,
    float* __restrict__ out)
{
  extern __shared__ char smem[];
  _Float16* Ps   = (_Float16*)(smem + SM_PS);   // p tile, 16 x 512 f16 (rows 10..15 zero)
  _Float16* Qs   = (_Float16*)(smem + SM_QS);   // Q, row-major f16
  _Float16* Ks   = (_Float16*)(smem + SM_KS);   // K, row-major f16
  _Float16* Vt   = (_Float16*)(smem + SM_VT);   // V transposed: [d][f], f16
  _Float16* Os   = (_Float16*)(smem + SM_OS);   // attention output, row-major f16
  _Float16* At   = (_Float16*)(smem + SM_AT);   // softmaxed attn, per head 16x16 f16
  float*    Ofin = (float*)(smem + SM_QS);      // overlays Qs+Ks: 16 x 512 f32
  float*    xv   = (float*)(smem + SM_XV);
  float*    mu   = (float*)(smem + SM_MU);
  float*    rs   = (float*)(smem + SM_RS);
  _Float16* zpad = (_Float16*)(smem + SM_ZP);   // 16 zero halves for padded B loads

  const int b    = blockIdx.x;
  const int t    = threadIdx.x;          // 0..127
  const int lane = t & 31;
  const int wave = t >> 5;               // 0..3, owns cols [wave*128, wave*128+128)
  const int nlo  = lane & 15;
  const int g    = lane >> 4;

  if (t < F_N) xv[t] = x[(size_t)b * F_N + t];
  if (t < 16)  zpad[t] = (_Float16)0.f;
  __syncthreads();

  // ---- stage 1: p[f][d] = relu(x[f]*Wf[f][d] + bf[f][d]), padded to 16 rows
  for (int f = 0; f < 16; ++f) {
    const float xf = (f < F_N) ? xv[f] : 0.f;
    for (int d = t; d < D_N; d += 128) {
      float v = 0.f;
      if (f < F_N) v = fmaxf(xf * Wf[f * D_N + d] + bf[f * D_N + d], 0.f);
      Ps[f * PS + d] = (_Float16)v;
    }
  }
  __syncthreads();

  // ---- stage 2: Q/K/V projections (y = p @ W^T + bias), WMMA 16x16x32
  for (int proj = 0; proj < 3; ++proj) {
    const _Float16* W  = Wh + proj * D_N * D_N;
    const float* bias  = (proj == 0) ? bq : (proj == 1) ? bk : bv;
    for (int nt = 0; nt < 8; ++nt) {
      const int n0 = (wave * 8 + nt) * 16;
      v8f acc = {};
      #pragma unroll 4
      for (int k0 = 0; k0 < D_N; k0 += 32) {
        v16h a  = lds_loadA(Ps, PS, k0, lane);
        v16h bm = glb_loadB(W, n0, k0, lane);
        acc = __builtin_amdgcn_wmma_f32_16x16x32_f16(false, a, false, bm,
                                                     (short)0, acc, false, false);
      }
      const float bb = bias[n0 + nlo];
      if (proj == 2) {           // store V transposed: Vt[d][f]
        #pragma unroll
        for (int r = 0; r < 8; ++r)
          Vt[(n0 + nlo) * VTS + (r + 8 * g)] = (_Float16)(acc[r] + bb);
      } else {
        _Float16* dst = (proj == 0) ? Qs : Ks;
        #pragma unroll
        for (int r = 0; r < 8; ++r)
          dst[(r + 8 * g) * PS + n0 + nlo] = (_Float16)(acc[r] + bb);
      }
    }
  }
  __syncthreads();

  // ---- stage 3: attention; wave w handles heads 2w, 2w+1 (its own Q columns)
  for (int hh = 0; hh < 2; ++hh) {
    const int h  = wave * 2 + hh;
    const int hc = h * DH_N;
    v8f s = {};
    #pragma unroll
    for (int k0 = 0; k0 < DH_N; k0 += 32) {
      v16h a  = lds_loadA(Qs, PS, hc + k0, lane);          // Q rows
      v16h bm = lds_loadB_rows(Ks, PS, hc + k0, lane);     // B(k=d,n=f2)=K[f2][d]
      s = __builtin_amdgcn_wmma_f32_16x16x32_f16(false, a, false, bm,
                                                 (short)0, s, false, false);
    }
    // scale, mask columns f2 >= 10, row softmax across 16 lanes (wave32 shfl)
    const bool colmask = (nlo >= F_N);
    float av[8];
    #pragma unroll
    for (int r = 0; r < 8; ++r) {
      float sv = colmask ? -1e30f : s[r] * 0.125f;   // 1/sqrt(DH)
      float mx = sv;
      for (int off = 1; off < 16; off <<= 1) mx = fmaxf(mx, __shfl_xor(mx, off, 32));
      float ev = __expf(sv - mx);
      float sm = ev;
      for (int off = 1; off < 16; off <<= 1) sm += __shfl_xor(sm, off, 32);
      av[r] = ev / sm;
    }
    #pragma unroll
    for (int r = 0; r < 8; ++r)
      At[h * 16 * ATS + (r + 8 * g) * ATS + nlo] = (_Float16)av[r];
    __syncthreads();   // uniform: every wave runs both hh iterations

    // A fragment of attn: lanes<16 hold K=0..15 (contiguous); lanes>=16 meet zero B
    V16U af;
    {
      const _Float16* p = At + h * 16 * ATS + nlo * ATS;
      af.h[0] = *(const v8h*)p;
      af.h[1] = *(const v8h*)(p + 8);
    }
    // O_h = attn @ V_h : B(k=f2, n=d) = Vt[d][f2]; lanes>=16 (K=16..31) read zeros
    #pragma unroll
    for (int nt = 0; nt < 4; ++nt) {
      const int d0 = hc + nt * 16;
      const _Float16* vp = (lane < 16) ? (Vt + (d0 + nlo) * VTS) : zpad;
      V16U bm;
      bm.h[0] = *(const v8h*)(vp);
      bm.h[1] = *(const v8h*)(vp + 8);
      v8f o = {};
      o = __builtin_amdgcn_wmma_f32_16x16x32_f16(false, af.v, false, bm.v,
                                                 (short)0, o, false, false);
      #pragma unroll
      for (int r = 0; r < 8; ++r)
        Os[(r + 8 * g) * PS + d0 + nlo] = (_Float16)o[r];
    }
  }
  __syncthreads();

  // ---- stage 4: output projection o2 = O @ Wo^T + bo -> f32 in LDS
  {
    const _Float16* W = Wh + 3 * D_N * D_N;
    for (int nt = 0; nt < 8; ++nt) {
      const int n0 = (wave * 8 + nt) * 16;
      v8f acc = {};
      #pragma unroll 4
      for (int k0 = 0; k0 < D_N; k0 += 32) {
        v16h a  = lds_loadA(Os, PS, k0, lane);
        v16h bm = glb_loadB(W, n0, k0, lane);
        acc = __builtin_amdgcn_wmma_f32_16x16x32_f16(false, a, false, bm,
                                                     (short)0, acc, false, false);
      }
      const float bb = bo[n0 + nlo];
      #pragma unroll
      for (int r = 0; r < 8; ++r)
        Ofin[(r + 8 * g) * OFS + n0 + nlo] = acc[r] + bb;
    }
  }
  __syncthreads();

  // ---- stage 5: LayerNorm stats per valid row (one wave per row, striped)
  for (int f = wave; f < F_N; f += 4) {
    float s1 = 0.f, s2 = 0.f;
    #pragma unroll 4
    for (int j = 0; j < 16; ++j) {
      float v = Ofin[f * OFS + lane * 16 + j];
      s1 += v; s2 += v * v;
    }
    for (int off = 16; off; off >>= 1) {
      s1 += __shfl_xor(s1, off, 32);
      s2 += __shfl_xor(s2, off, 32);
    }
    if (lane == 0) {
      const float m = s1 * (1.f / D_N);
      const float var = s2 * (1.f / D_N) - m * m;
      mu[f] = m;
      rs[f] = rsqrtf(var + 1e-5f);
    }
  }
  __syncthreads();

  // ---- stage 6: out[b][d] = gamma[d]*sum_f (o-mu_f)*r_f + F*beta[d]
  for (int d = t; d < D_N; d += 128) {
    float acc = 0.f;
    #pragma unroll
    for (int f = 0; f < F_N; ++f)
      acc += (Ofin[f * OFS + d] - mu[f]) * rs[f];
    out[(size_t)b * D_N + d] = acc * gamma[d] + (float)F_N * beta[d];
  }
}

// ------------------------------ launcher -----------------------------------
extern "C" void kernel_launch(void* const* d_in, const int* in_sizes, int n_in,
                              void* d_out, int out_size, void* d_ws, size_t ws_size,
                              hipStream_t stream) {
  (void)in_sizes; (void)n_in; (void)out_size; (void)ws_size;
  const float* x     = (const float*)d_in[0];
  const float* Wf    = (const float*)d_in[1];
  const float* bf    = (const float*)d_in[2];
  const float* Wq    = (const float*)d_in[3];
  const float* Wk    = (const float*)d_in[4];
  const float* Wv    = (const float*)d_in[5];
  const float* bq    = (const float*)d_in[6];
  const float* bk    = (const float*)d_in[7];
  const float* bv    = (const float*)d_in[8];
  const float* Wo    = (const float*)d_in[9];
  const float* bo    = (const float*)d_in[10];
  const float* gamma = (const float*)d_in[11];
  const float* beta  = (const float*)d_in[12];
  float* out = (float*)d_out;

  _Float16* Wh = (_Float16*)d_ws;   // 4 * 512*512 f16 = 2 MB scratch

  convert_weights_kernel<<<dim3((D_N * D_N) / 256), dim3(256), 0, stream>>>(
      Wq, Wk, Wv, Wo, Wh);

  fused_attn_kernel<<<dim3(B_N), dim3(128), SMEM_BYTES, stream>>>(
      x, Wf, bf, Wh, bq, bk, bv, bo, gamma, beta, out);
}
